// SelfAttention_PreLN_24026047054582
// MI455X (gfx1250) — matrix operand order repaired and checked by reading, not compiled, not run on hardware
//
#include <hip/hip_runtime.h>

typedef __attribute__((ext_vector_type(16))) __bf16 v16bf;
typedef __attribute__((ext_vector_type(8)))  __bf16 v8bf;
typedef __attribute__((ext_vector_type(8)))  float  v8f;

#define B_   2
#define L_   2048
#define C_   1024
#define H_   16
#define DK_  64
#define NQT  (L_ / 16)          // 128 q-tiles per (b,h)
#define MASKV (-32767.0f)

// ---------------------------------------------------------------------------
// CDNA5 primitives
// ---------------------------------------------------------------------------
__device__ __forceinline__ v8f wmma_bf16(v16bf a, v16bf b, v8f c) {
  return __builtin_amdgcn_wmma_f32_16x16x32_bf16(false, a, false, b,
                                                 (short)0, c, false, false);
}

// async copy of 16 bytes global -> LDS, tracked by ASYNCcnt (GV addressing)
__device__ __forceinline__ void async_b128(const void* gsrc, const void* ldst) {
  uint64_t src = (uint64_t)(uintptr_t)gsrc;
  uint32_t dst = (uint32_t)(uintptr_t)ldst;   // shared aperture: LDS off in [31:0]
  asm volatile("global_load_async_to_lds_b128 %0, %1, off"
               :: "v"(dst), "v"(src) : "memory");
}
__device__ __forceinline__ void wait_async_le2() {
  asm volatile("s_wait_asynccnt 0x2" ::: "memory");
}
__device__ __forceinline__ void wait_async_le4() {
  asm volatile("s_wait_asynccnt 0x4" ::: "memory");
}
__device__ __forceinline__ void wait_async_0() {
  asm volatile("s_wait_asynccnt 0x0" ::: "memory");
}

// B-fragment: 16 contiguous bf16 per lane (k = e + 16*(lane>=16) via pointer)
__device__ __forceinline__ v16bf ld_bfrag(const __bf16* p) {
  v8bf lo = *(const v8bf*)(p);
  v8bf hi = *(const v8bf*)(p + 8);
  return __builtin_shufflevector(lo, hi, 0, 1, 2, 3, 4, 5, 6, 7,
                                 8, 9, 10, 11, 12, 13, 14, 15);
}
// A-fragment: 16 contiguous bf16 per lane (k = e + 16*(lane>=16) via pointer)
__device__ __forceinline__ v16bf ld_afrag(const __bf16* p) {
  v8bf lo = *(const v8bf*)(p);
  v8bf hi = *(const v8bf*)(p + 8);
  return __builtin_shufflevector(lo, hi, 0, 1, 2, 3, 4, 5, 6, 7,
                                 8, 9, 10, 11, 12, 13, 14, 15);
}

// ---------------------------------------------------------------------------
// Weight convert + transpose: src fp32 [K=1024][N=1024] -> dst bf16 [N][K]
// ---------------------------------------------------------------------------
__global__ __launch_bounds__(256)
void cvtT_kernel(const float* __restrict__ src, __bf16* __restrict__ dst) {
  int i = blockIdx.x * 256 + threadIdx.x;   // over 1<<20 elements
  int k = i >> 10;
  int n = i & (C_ - 1);
  dst[(size_t)n * C_ + k] = (__bf16)src[i];
}

// ---------------------------------------------------------------------------
// Fused LayerNorm -> bf16.  grid = (B*L, 3), block = 128
// ---------------------------------------------------------------------------
__global__ __launch_bounds__(128)
void ln_kernel(const float* __restrict__ qx, const float* __restrict__ kx,
               const float* __restrict__ vx, const float* __restrict__ g,
               const float* __restrict__ bet,
               __bf16* __restrict__ qn, __bf16* __restrict__ kn,
               __bf16* __restrict__ vn) {
  __shared__ float reds[2][4];
  int row   = blockIdx.x;
  int which = blockIdx.y;
  const float* src = (which == 0 ? qx : which == 1 ? kx : vx) + (size_t)row * C_;
  __bf16*      dst = (which == 0 ? qn : which == 1 ? kn : vn) + (size_t)row * C_;
  int tid = threadIdx.x;

  float4 x0 = ((const float4*)src)[tid * 2 + 0];
  float4 x1 = ((const float4*)src)[tid * 2 + 1];
  float xs[8] = {x0.x, x0.y, x0.z, x0.w, x1.x, x1.y, x1.z, x1.w};
  float s = 0.f, ss = 0.f;
#pragma unroll
  for (int j = 0; j < 8; ++j) { s += xs[j]; ss += xs[j] * xs[j]; }
#pragma unroll
  for (int m = 16; m >= 1; m >>= 1) {
    s  += __shfl_xor(s,  m, 32);
    ss += __shfl_xor(ss, m, 32);
  }
  if ((tid & 31) == 0) { reds[0][tid >> 5] = s; reds[1][tid >> 5] = ss; }
  __syncthreads();
  s  = reds[0][0] + reds[0][1] + reds[0][2] + reds[0][3];
  ss = reds[1][0] + reds[1][1] + reds[1][2] + reds[1][3];
  float mean = s * (1.0f / C_);
  float var  = ss * (1.0f / C_) - mean * mean;
  float rstd = rsqrtf(var + 1e-5f);
  int base = tid * 8;
#pragma unroll
  for (int j = 0; j < 8; ++j)
    dst[base + j] = (__bf16)((xs[j] - mean) * rstd * g[base + j] + bet[base + j]);
}

// ---------------------------------------------------------------------------
// Wave-tile GEMM with async A-panel staging.
// Block = 8 waves; all waves share the same 16-row M tile (tm uniform/block).
// A K-panel staged in 8KB chunks (16 x 256 bf16), double-buffered via
// GLOBAL_LOAD_ASYNC_TO_LDS_B128 + s_wait_asynccnt.  W columns stream directly
// (unique per wave, contiguous b128).
// mode 0: bf16 [B,H,L,DK] | mode 1: bf16 [B,H,DK,L] | mode 2: fp32 [M,1024]
// ---------------------------------------------------------------------------
__global__ __launch_bounds__(256)
void gemm_kernel(const __bf16* __restrict__ A, const __bf16* __restrict__ Wt,
                 const float* __restrict__ bias, void* __restrict__ out,
                 int mode) {
  __shared__ __bf16 abuf[2][16][256];      // 2 x 8KB
  int tid  = threadIdx.x;
  int lane = tid & 31;
  int half = lane >> 4, ln = lane & 15;
  int gw = blockIdx.x * 8 + (tid >> 5);
  const int TN = C_ / 64;                  // 16 column tiles
  int tm = gw / TN, tn = gw % TN;          // tm uniform within block

  const __bf16* Ablk = A + (size_t)(tm * 16) * C_;
  const __bf16* wcol = Wt + (size_t)(tn * 64 + ln) * C_ + half * 16;

  // stage chunk c (k in [c*256, c*256+256)) into abuf[bufi]: 2 b128/thread
  auto stage = [&](int c, int bufi) {
#pragma unroll
    for (int u = 0; u < 2; ++u) {
      int cc  = tid * 2 + u;               // 512 chunks of 8 elements
      int row = cc >> 5;
      int ko  = (cc & 31) * 8;
      async_b128(Ablk + (size_t)row * C_ + c * 256 + ko, &abuf[bufi][row][ko]);
    }
  };

  stage(0, 0);
  v8f acc0 = {}, acc1 = {}, acc2 = {}, acc3 = {};
  for (int c = 0; c < 4; ++c) {
    if (c < 3) { stage(c + 1, (c + 1) & 1); wait_async_le2(); }
    else       { wait_async_0(); }
    __syncthreads();                        // staged panel visible to all waves
    __builtin_prefetch(wcol + (c + 1) * 256, 0, 1);
    const __bf16* arow = &abuf[c & 1][ln][half * 16];
#pragma unroll
    for (int kk = 0; kk < 256; kk += 32) {
      int k0 = c * 256 + kk;
      v16bf af = ld_afrag(arow + kk);
      acc0 = wmma_bf16(af, ld_bfrag(wcol + k0),           acc0);
      acc1 = wmma_bf16(af, ld_bfrag(wcol + 16 * C_ + k0), acc1);
      acc2 = wmma_bf16(af, ld_bfrag(wcol + 32 * C_ + k0), acc2);
      acc3 = wmma_bf16(af, ld_bfrag(wcol + 48 * C_ + k0), acc3);
    }
    __syncthreads();                        // panel consumed before overwrite
  }

#pragma unroll
  for (int t = 0; t < 4; ++t) {
    v8f acc = (t == 0) ? acc0 : (t == 1) ? acc1 : (t == 2) ? acc2 : acc3;
#pragma unroll
    for (int v = 0; v < 8; ++v) {
      int m = tm * 16 + v + 8 * half;
      int n = tn * 64 + t * 16 + ln;
      float val = acc[v] + bias[n];
      if (mode == 2) {
        ((float*)out)[(size_t)m * C_ + n] = val;
      } else {
        int b = m >> 11, l = m & (L_ - 1);
        int h = n >> 6,  d = n & (DK_ - 1);
        if (mode == 0)
          ((__bf16*)out)[(((size_t)(b * H_ + h)) * L_ + l) * DK_ + d] = (__bf16)val;
        else
          ((__bf16*)out)[(((size_t)(b * H_ + h)) * DK_ + d) * L_ + l] = (__bf16)val;
      }
    }
  }
}

// ---------------------------------------------------------------------------
// Flash attention, async-staged K/V tiles.
// Block = 4 waves sharing one (b,h); each wave owns a 16-query strip.
// Per 32-key step: K tile (32x64) + V tile (64x32) staged once per block into
// LDS (double-buffered, 4 async b128/thread), then all 4 waves feed WMMA from
// ds_load_b128 — 4x less global traffic than per-wave streaming.
// ---------------------------------------------------------------------------
__global__ __launch_bounds__(128)
void attn_kernel(const __bf16* __restrict__ Q, const __bf16* __restrict__ K,
                 const __bf16* __restrict__ Vt, const int* __restrict__ maskPAD,
                 __bf16* __restrict__ Z) {
  __shared__ __bf16 kbuf[2][32][64];    // 2 x 4KB, row = key, contiguous dk
  __shared__ __bf16 vbuf[2][64][32];    // 2 x 4KB, row = feature, contiguous keys
  __shared__ __bf16 plds[4][16][32];    // probs C-layout -> A-layout bounce
  int tid  = threadIdx.x;
  int lane = tid & 31;
  int wid  = tid >> 5;
  int half = lane >> 4, ln = lane & 15;
  int gw = blockIdx.x * 4 + wid;
  int qt = gw & (NQT - 1);
  int h  = (gw >> 7) & (H_ - 1);        // uniform within block
  int b  = gw >> 11;

  const __bf16* Qb = Q  + ((size_t)(b * H_ + h) * L_ + qt * 16) * DK_;
  const __bf16* Kb = K  + (size_t)(b * H_ + h) * L_ * DK_;
  const __bf16* Vb = Vt + (size_t)(b * H_ + h) * DK_ * L_;
  const int*    Mb = maskPAD + (size_t)b * L_ * L_;

  // cooperative async stage of tile ti (keys [ti*32, ti*32+32))
  auto stage = [&](int ti) {
    int bufi = ti & 1;
    int kb   = ti * 32;
    // K tile: 2048 contiguous elements; thread t -> elements [16t, 16t+16)
    int e0 = tid * 16;
    async_b128(Kb + (size_t)kb * DK_ + e0,     &kbuf[bufi][0][0] + e0);
    async_b128(Kb + (size_t)kb * DK_ + e0 + 8, &kbuf[bufi][0][0] + e0 + 8);
    // V tile: 64 rows x 32 elems; 256 chunks of 8; thread t -> chunks 2t,2t+1
#pragma unroll
    for (int u = 0; u < 2; ++u) {
      int cc  = tid * 2 + u;
      int row = cc >> 2;
      int off = (cc & 3) * 8;
      async_b128(Vb + (size_t)row * L_ + kb + off, &vbuf[bufi][row][off]);
    }
  };

  const __bf16* qp = Qb + ln * DK_ + half * 16;
  v16bf qa0 = ld_afrag(qp);        // dk 0..31
  v16bf qa1 = ld_afrag(qp + 32);   // dk 32..63

  v8f z0 = {}, z1 = {}, z2 = {}, z3 = {};
  float rmax[8], rsum[8];
#pragma unroll
  for (int v = 0; v < 8; ++v) { rmax[v] = -3.0e38f; rsum[v] = 0.f; }

  stage(0);
  for (int ti = 0; ti < L_ / 32; ++ti) {
    int kb   = ti * 32;
    int bufi = ti & 1;
    if (ti < L_ / 32 - 1) { stage(ti + 1); wait_async_le4(); }
    else                  { wait_async_0(); }
    __syncthreads();                          // K/V tiles visible to all waves

    // ---- scores: two 16x16 tiles over DK=64 (2 WMMAs each), from LDS ----
    v8f s0 = {}, s1 = {};
    {
      const __bf16* kp = &kbuf[bufi][ln][half * 16];
      s0 = wmma_bf16(qa0, ld_bfrag(kp),      s0);
      s0 = wmma_bf16(qa1, ld_bfrag(kp + 32), s0);
    }
    {
      const __bf16* kp = &kbuf[bufi][16 + ln][half * 16];
      s1 = wmma_bf16(qa0, ld_bfrag(kp),      s1);
      s1 = wmma_bf16(qa1, ld_bfrag(kp + 32), s1);
    }
    // ---- scale, mask, online softmax stats (fp32) ----
    float corr[8], p0[8], p1[8];
#pragma unroll
    for (int v = 0; v < 8; ++v) {
      int q = qt * 16 + v + 8 * half;
      int mk0 = Mb[(size_t)q * L_ + kb + ln];
      int mk1 = Mb[(size_t)q * L_ + kb + 16 + ln];
      float a0 = mk0 ? s0[v] * 0.125f : MASKV;
      float a1 = mk1 ? s1[v] * 0.125f : MASKV;
      float t = fmaxf(a0, a1);
      t = fmaxf(t, __shfl_xor(t, 1, 16));
      t = fmaxf(t, __shfl_xor(t, 2, 16));
      t = fmaxf(t, __shfl_xor(t, 4, 16));
      t = fmaxf(t, __shfl_xor(t, 8, 16));
      float mnew = fmaxf(rmax[v], t);
      corr[v] = __expf(rmax[v] - mnew);
      rmax[v] = mnew;
      p0[v] = __expf(a0 - mnew);
      p1[v] = __expf(a1 - mnew);
      float ts = p0[v] + p1[v];
      ts += __shfl_xor(ts, 1, 16);
      ts += __shfl_xor(ts, 2, 16);
      ts += __shfl_xor(ts, 4, 16);
      ts += __shfl_xor(ts, 8, 16);
      rsum[v] = rsum[v] * corr[v] + ts;
    }
#pragma unroll
    for (int v = 0; v < 8; ++v) {
      z0[v] *= corr[v]; z1[v] *= corr[v]; z2[v] *= corr[v]; z3[v] *= corr[v];
    }
    // ---- C-layout probs -> LDS -> A-layout fragment ----
#pragma unroll
    for (int v = 0; v < 8; ++v) {
      plds[wid][v + 8 * half][ln]      = (__bf16)p0[v];
      plds[wid][v + 8 * half][16 + ln] = (__bf16)p1[v];
    }
    __syncthreads();
    v16bf af = ld_afrag(&plds[wid][ln][half * 16]);
    // ---- Z += alpha(16x32) @ V(32x64): 4 WMMAs from LDS V tile ----
    z0 = wmma_bf16(af, ld_bfrag(&vbuf[bufi][ 0 + ln][half * 16]), z0);
    z1 = wmma_bf16(af, ld_bfrag(&vbuf[bufi][16 + ln][half * 16]), z1);
    z2 = wmma_bf16(af, ld_bfrag(&vbuf[bufi][32 + ln][half * 16]), z2);
    z3 = wmma_bf16(af, ld_bfrag(&vbuf[bufi][48 + ln][half * 16]), z3);
    __syncthreads();                 // tiles/plds consumed before next overwrite
  }
  // ---- normalize and store Z as [B, L, H*DK] bf16 ----
#pragma unroll
  for (int v = 0; v < 8; ++v) {
    float inv = 1.0f / rsum[v];
    int q = qt * 16 + v + 8 * half;
    size_t rowoff = ((size_t)(b * L_ + q)) * (H_ * DK_) + h * DK_;
    Z[rowoff +  0 + ln] = (__bf16)(z0[v] * inv);
    Z[rowoff + 16 + ln] = (__bf16)(z1[v] * inv);
    Z[rowoff + 32 + ln] = (__bf16)(z2[v] * inv);
    Z[rowoff + 48 + ln] = (__bf16)(z3[v] * inv);
  }
}

// ---------------------------------------------------------------------------
// Host side
// ---------------------------------------------------------------------------
extern "C" void kernel_launch(void* const* d_in, const int* in_sizes, int n_in,
                              void* d_out, int out_size, void* d_ws,
                              size_t ws_size, hipStream_t stream) {
  (void)in_sizes; (void)n_in; (void)out_size; (void)ws_size;
  const float* qx   = (const float*)d_in[0];
  const float* kx   = (const float*)d_in[1];
  const float* vx   = (const float*)d_in[2];
  const int*   mask = (const int*)  d_in[3];
  const float* lng  = (const float*)d_in[4];
  const float* lnb  = (const float*)d_in[5];
  const float* wq   = (const float*)d_in[6];
  const float* bq   = (const float*)d_in[7];
  const float* wk   = (const float*)d_in[8];
  const float* bk   = (const float*)d_in[9];
  const float* wv   = (const float*)d_in[10];
  const float* bv   = (const float*)d_in[11];
  const float* wo   = (const float*)d_in[12];
  const float* bo   = (const float*)d_in[13];

  char* ws = (char*)d_ws;
  const size_t MB = 1u << 20;
  __bf16* qn  = (__bf16*)(ws + 0 * MB);    // 8 MB each
  __bf16* kn  = (__bf16*)(ws + 8 * MB);
  __bf16* vn  = (__bf16*)(ws + 16 * MB);
  __bf16* wqb = (__bf16*)(ws + 24 * MB);   // 2 MB each, [N][K]
  __bf16* wkb = (__bf16*)(ws + 26 * MB);
  __bf16* wvb = (__bf16*)(ws + 28 * MB);
  __bf16* wob = (__bf16*)(ws + 30 * MB);
  __bf16* Qp  = (__bf16*)(ws + 32 * MB);   // [B,H,L,DK]
  __bf16* Kp  = (__bf16*)(ws + 40 * MB);   // [B,H,L,DK]
  __bf16* Vp  = (__bf16*)(ws + 48 * MB);   // [B,H,DK,L]
  __bf16* Zp  = (__bf16*)(ws + 0 * MB);    // reuse qn region (dead by then)

  // 1) weights -> bf16, transposed [N][K]
  cvtT_kernel<<<4096, 256, 0, stream>>>(wq, wqb);
  cvtT_kernel<<<4096, 256, 0, stream>>>(wk, wkb);
  cvtT_kernel<<<4096, 256, 0, stream>>>(wv, wvb);
  cvtT_kernel<<<4096, 256, 0, stream>>>(wo, wob);
  // 2) fused LayerNorm -> bf16
  ln_kernel<<<dim3(B_ * L_, 3), 128, 0, stream>>>(qx, kx, vx, lng, lnb,
                                                  qn, kn, vn);
  // 3) projections (WMMA GEMMs, async A staging)
  gemm_kernel<<<512, 256, 0, stream>>>(qn, wqb, bq, (void*)Qp, 0);
  gemm_kernel<<<512, 256, 0, stream>>>(kn, wkb, bk, (void*)Kp, 0);
  gemm_kernel<<<512, 256, 0, stream>>>(vn, wvb, bv, (void*)Vp, 1);
  // 4) flash attention (async K/V staging + WMMA)
  attn_kernel<<<1024, 128, 0, stream>>>(Qp, Kp, Vp, mask, Zp);
  // 5) output projection, fp32 epilogue into d_out
  gemm_kernel<<<512, 256, 0, stream>>>(Zp, wob, bo, d_out, 2);
}
